// FlowLSTM_75479755260061
// MI455X (gfx1250) — matrix operand order repaired and checked by reading, not compiled, not run on hardware
//
#include <hip/hip_runtime.h>
#include <hip/hip_bf16.h>

// CDNA5 / gfx1250 fused LSTM. wave32, WMMA f16 -> f32, async global->LDS staging.
typedef __attribute__((ext_vector_type(16))) _Float16 v16h;
typedef __attribute__((ext_vector_type(8)))  float    v8f;

#define LSTM_I 17
#define LSTM_H 128
#define LSTM_T 19
#define LSTM_B 32768
#define BM 16      // batch rows per workgroup (one WMMA M-tile)
#define NT 512     // threads per workgroup = 16 wave32

// Single-instruction v_exp / v_rcp (TRANS unit, co-executes with WMMA pipe).
// sig: exp->inf => rcp(inf)=0; exp->0 => rcp(1)=1  (no NaN at extremes)
__device__ __forceinline__ float sigf(float x) {
    return __builtin_amdgcn_rcpf(1.0f + __expf(-x));
}
// tanh(x) = 1 - 2/(exp(2x)+1): exp->inf => 1; exp->0 => -1  (no NaN at extremes)
__device__ __forceinline__ float tanh_fast(float x) {
    return 1.0f - 2.0f * __builtin_amdgcn_rcpf(__expf(2.0f * x) + 1.0f);
}

// Build a 16x32 f16 A-operand fragment from a row-major f16 LDS tile.
// K order used here (contiguous, not AMD table): lane = (khalf<<4)|M ; elements 0..15 -> K = kb16..kb16+15,
// kb16 = (lane>>4)*16, all + kbase (matches B-operand K order).
__device__ __forceinline__ v16h load_afrag(const _Float16* __restrict__ buf,
                                           int stride, int kbase, int lane) {
    const int M   = lane & 15;
    const int off = kbase + ((lane >> 4) << 4);
    const _Float16* p = buf + M * stride + off;
    v16h a;
#pragma unroll
    for (int e = 0; e < 16; ++e) { a[e] = p[e]; }
    return a;
}

// Same fragment, but from an f32 LDS tile (converts while packing).
__device__ __forceinline__ v16h load_afrag_f32(const float* __restrict__ buf,
                                               int stride, int lane) {
    const int M   = lane & 15;
    const int off = (lane >> 4) << 4;
    const float* p = buf + M * stride + off;
    v16h a;
#pragma unroll
    for (int e = 0; e < 16; ++e) {
        a[e]     = (_Float16)p[e];
    }
    return a;
}

__global__ void __launch_bounds__(NT)
flow_lstm_kernel(const float* __restrict__ x,      // [B,T,I]
                 const float* __restrict__ W_ih,   // [4H,I]
                 const float* __restrict__ W_hh,   // [4H,H]
                 const float* __restrict__ b_ih,   // [4H]
                 const float* __restrict__ b_hh,   // [4H]
                 const float* __restrict__ W_out,  // [I,H]
                 const float* __restrict__ b_out,  // [I]
                 float* __restrict__ out)          // [B,T,I]
{
    __shared__ float    xbuf[BM * 32];            // x_t staged f32, K padded 17->32 (2 KB)
    __shared__ _Float16 hbuf[BM * LSTM_H];        // h state, f16 (4 KB)
    __shared__ float    gbuf[BM * 4 * LSTM_H];    // gates (32 KB)

    const int tid  = threadIdx.x;
    const int lane = tid & 31;
    const int wave = tid >> 5;            // 0..15
    const int nlo  = lane & 15;
    const int row0 = blockIdx.x * BM;

    // ---- Preload weight B-operand fragments into VGPRs (reused for all 19 steps).
    // B-operand K order used here (contiguous, same as A): lane (khalf<<4)|N ; elements 0..15 -> K = khalf*16 + e.
    v16h  bh[2][4];   // W_hh: wave covers gate columns [wave*32, wave*32+32): 2 N-tiles x 4 K-blocks
    v16h  bi[2];      // W_ih: single K-block (17 padded to 32)
    float bias[2];
    const int kb16 = (lane >> 4) << 4;
#pragma unroll
    for (int tt = 0; tt < 2; ++tt) {
        const int n = wave * 32 + tt * 16 + nlo;   // gate column 0..511
        bias[tt] = b_ih[n] + b_hh[n];
#pragma unroll
        for (int q = 0; q < 4; ++q) {
            const float* wr = W_hh + n * LSTM_H + q * 32 + kb16;
#pragma unroll
            for (int e = 0; e < 16; ++e) bh[tt][q][e] = (_Float16)wr[e];
        }
#pragma unroll
        for (int e = 0; e < 16; ++e) {
            const int k = kb16 + e;
            bi[tt][e] = (k < LSTM_I) ? (_Float16)W_ih[n * LSTM_I + k] : (_Float16)0.0f;
        }
    }

    // Output-projection fragments for waves 0,1 (N = output col, padded 17->32).
    v16h  bo[4];
    float boutv = 0.0f;
    if (wave < 2) {
        const int n = wave * 16 + nlo;             // 0..31, valid < 17
        boutv = (n < LSTM_I) ? b_out[n] : 0.0f;
#pragma unroll
        for (int q = 0; q < 4; ++q) {
#pragma unroll
            for (int e = 0; e < 16; ++e)
                bo[q][e] = (n < LSTM_I) ? (_Float16)W_out[n * LSTM_H + q * 32 + kb16 + e]
                                        : (_Float16)0.0f;
        }
    }

    // ---- init state: c in registers, h = 0 in LDS, xbuf zero (pad cols stay 0;
    // async copies only ever touch cols < 17).
    float creg[4] = {0.0f, 0.0f, 0.0f, 0.0f};
#pragma unroll
    for (int k = 0; k < 4; ++k) hbuf[tid * 4 + k] = (_Float16)0.0f;
    xbuf[tid] = 0.0f;

    const int xr = tid >> 5, xc = tid & 31;  // x staging: one elem/thread (16x32)
    const int er = tid >> 5;                 // elementwise row
    const int en = (tid & 31) * 4;           // elementwise col base

    // LDS byte offset of this thread's xbuf slot (flat->LDS truncation: low 32
    // bits of a generic pointer to __shared__ are the LDS offset).
    const unsigned xbuf_off =
        (unsigned)(uintptr_t)(&xbuf[xr * 32 + xc]);

    for (int t = 0; t < LSTM_T; ++t) {
        // Phase A: async copy x_t global -> LDS (no VGPR round-trip, ASYNCcnt).
        if (xc < LSTM_I) {
            unsigned long long gaddr = (unsigned long long)(uintptr_t)
                (x + ((size_t)(row0 + xr) * LSTM_T + t) * LSTM_I + xc);
            asm volatile("global_load_async_to_lds_b32 %0, %1, off"
                         :: "v"(xbuf_off), "v"(gaddr) : "memory");
        }
        asm volatile("s_wait_asynccnt 0x0" ::: "memory");
        __syncthreads();

        // Phase B: gates = (b_ih+b_hh) + x_t @ W_ih^T + h @ W_hh^T  (WMMA)
        {
            // Issue ALL A-fragment LDS loads up front (distinct live regs) so the
            // dscnt waits count down while the WMMA chain drains: no serialized
            // LDS round-trips between WMMA pairs.
            v16h ax  = load_afrag_f32(xbuf, 32, lane);
            v16h ah0 = load_afrag(hbuf, LSTM_H,  0, lane);
            v16h ah1 = load_afrag(hbuf, LSTM_H, 32, lane);
            v16h ah2 = load_afrag(hbuf, LSTM_H, 64, lane);
            v16h ah3 = load_afrag(hbuf, LSTM_H, 96, lane);

            v8f acc0, acc1;
#pragma unroll
            for (int e = 0; e < 8; ++e) { acc0[e] = bias[0]; acc1[e] = bias[1]; }
            acc0 = __builtin_amdgcn_wmma_f32_16x16x32_f16(false, ax,  false, bi[0],
                                                          (short)0, acc0, false, false);
            acc1 = __builtin_amdgcn_wmma_f32_16x16x32_f16(false, ax,  false, bi[1],
                                                          (short)0, acc1, false, false);
            acc0 = __builtin_amdgcn_wmma_f32_16x16x32_f16(false, ah0, false, bh[0][0],
                                                          (short)0, acc0, false, false);
            acc1 = __builtin_amdgcn_wmma_f32_16x16x32_f16(false, ah0, false, bh[1][0],
                                                          (short)0, acc1, false, false);
            acc0 = __builtin_amdgcn_wmma_f32_16x16x32_f16(false, ah1, false, bh[0][1],
                                                          (short)0, acc0, false, false);
            acc1 = __builtin_amdgcn_wmma_f32_16x16x32_f16(false, ah1, false, bh[1][1],
                                                          (short)0, acc1, false, false);
            acc0 = __builtin_amdgcn_wmma_f32_16x16x32_f16(false, ah2, false, bh[0][2],
                                                          (short)0, acc0, false, false);
            acc1 = __builtin_amdgcn_wmma_f32_16x16x32_f16(false, ah2, false, bh[1][2],
                                                          (short)0, acc1, false, false);
            acc0 = __builtin_amdgcn_wmma_f32_16x16x32_f16(false, ah3, false, bh[0][3],
                                                          (short)0, acc0, false, false);
            acc1 = __builtin_amdgcn_wmma_f32_16x16x32_f16(false, ah3, false, bh[1][3],
                                                          (short)0, acc1, false, false);

            // C/D layout: VGPR j, lane -> row M = j + 8*(lane>=16), col N = nlo
            const int rj = (lane >> 4) << 3;
#pragma unroll
            for (int j = 0; j < 8; ++j) {
                gbuf[(j + rj) * 512 + wave * 32 + nlo]      = acc0[j];
                gbuf[(j + rj) * 512 + wave * 32 + 16 + nlo] = acc1[j];
            }
        }
        __syncthreads();

        // Phase C: cell update. c stays in registers; h -> LDS as f16.
        // 5 v_exp + 5 v_rcp per cell (TRANS pipe), no IEEE divide sequences.
#pragma unroll
        for (int k = 0; k < 4; ++k) {
            const int n = en + k;
            const float ig = gbuf[er * 512 + n];
            const float fg = gbuf[er * 512 + n + 128];
            const float gg = gbuf[er * 512 + n + 256];
            const float og = gbuf[er * 512 + n + 384];
            const float cv = sigf(fg) * creg[k] + sigf(ig) * tanh_fast(gg);
            creg[k] = cv;
            hbuf[er * LSTM_H + n] = (_Float16)(sigf(og) * tanh_fast(cv));
        }
        __syncthreads();

        // Phase D: y_t = h @ W_out^T + b_out (waves 0,1), store straight to HBM.
        if (wave < 2) {
            v16h ah0 = load_afrag(hbuf, LSTM_H,  0, lane);
            v16h ah1 = load_afrag(hbuf, LSTM_H, 32, lane);
            v16h ah2 = load_afrag(hbuf, LSTM_H, 64, lane);
            v16h ah3 = load_afrag(hbuf, LSTM_H, 96, lane);
            v8f yacc;
#pragma unroll
            for (int e = 0; e < 8; ++e) yacc[e] = boutv;
            yacc = __builtin_amdgcn_wmma_f32_16x16x32_f16(false, ah0, false, bo[0],
                                                          (short)0, yacc, false, false);
            yacc = __builtin_amdgcn_wmma_f32_16x16x32_f16(false, ah1, false, bo[1],
                                                          (short)0, yacc, false, false);
            yacc = __builtin_amdgcn_wmma_f32_16x16x32_f16(false, ah2, false, bo[2],
                                                          (short)0, yacc, false, false);
            yacc = __builtin_amdgcn_wmma_f32_16x16x32_f16(false, ah3, false, bo[3],
                                                          (short)0, yacc, false, false);
            const int col = wave * 16 + nlo;
            if (col < LSTM_I) {
                const int rj = (lane >> 4) << 3;
#pragma unroll
                for (int j = 0; j < 8; ++j)
                    out[((size_t)(row0 + j + rj) * LSTM_T + t) * LSTM_I + col] = yacc[j];
            }
        }
        // No trailing barrier needed: next Phase A only writes xbuf (async), and
        // the barrier after it orders Phase D's hbuf reads before Phase C rewrites.
    }
}

extern "C" void kernel_launch(void* const* d_in, const int* in_sizes, int n_in,
                              void* d_out, int out_size, void* d_ws, size_t ws_size,
                              hipStream_t stream) {
    const float* x     = (const float*)d_in[0];
    const float* W_ih  = (const float*)d_in[1];
    const float* W_hh  = (const float*)d_in[2];
    const float* b_ih  = (const float*)d_in[3];
    const float* b_hh  = (const float*)d_in[4];
    const float* W_out = (const float*)d_in[5];
    const float* b_out = (const float*)d_in[6];
    float* out = (float*)d_out;

    dim3 grid(LSTM_B / BM);   // 2048 batch tiles
    dim3 block(NT);           // 16 wave32
    flow_lstm_kernel<<<grid, block, 0, stream>>>(x, W_ih, W_hh, b_ih, b_hh,
                                                 W_out, b_out, out);
}
